// MultiHeadAttention_67224828117347
// MI455X (gfx1250) — compile-verified, matrix operand order checked by reading, not run on hardware
//
#include <hip/hip_runtime.h>
#include <hip/hip_bf16.h>

typedef __attribute__((ext_vector_type(16))) __bf16 v16bf;
typedef __attribute__((ext_vector_type(8)))  __bf16 bf16x8;
typedef __attribute__((ext_vector_type(8)))  float  v8f;

#define SEQ 2048
#define DMODEL 1024
#define NHEAD 16
#define HDIM 64
#define NTOK 4096   // B*S

// ---- helpers ---------------------------------------------------------------

__device__ __forceinline__ v8f wmma_bf16(v16bf a, v16bf b, v8f c) {
  return __builtin_amdgcn_wmma_f32_16x16x32_bf16(false, a, false, b, (short)0, c,
                                                 false, false);
}

// Load one 16x32 bf16 A/B fragment row for this lane (two ds/global b128).
__device__ __forceinline__ v16bf load_frag(const __bf16* base, int half) {
  union { v16bf v; bf16x8 h[2]; } u;
  u.h[0] = *(const bf16x8*)(base + 8 * half);
  u.h[1] = *(const bf16x8*)(base + 16 + 8 * half);
  return u.v;
}

// Async copy 16B global -> LDS (per lane), tracked by ASYNCcnt.
__device__ __forceinline__ void async_copy16(void* lds_ptr, const void* gptr) {
  unsigned loff = (unsigned)(uintptr_t)lds_ptr;  // generic ptr low 32b = LDS offset
  unsigned long long ga = (unsigned long long)(uintptr_t)gptr;
  asm volatile("global_load_async_to_lds_b128 %0, %1, off"
               :: "v"(loff), "v"(ga) : "memory");
}

#define WAIT_ASYNC(n) asm volatile("s_wait_asynccnt " #n ::: "memory")
#define WAIT_DS0()    asm volatile("s_wait_dscnt 0x0" ::: "memory")

// xor-shuffle within 16-lane groups via ds_swizzle (group-of-32, and_mask=0x1f)
#define SWZ_XOR(v, m) \
  __int_as_float(__builtin_amdgcn_ds_swizzle(__float_as_int(v), (((m) << 10) | 0x1f)))

// ---- conversion kernels ----------------------------------------------------

__global__ void cvt_f32_bf16_kernel(const float* __restrict__ in,
                                    __bf16* __restrict__ out, int n) {
  int i = (blockIdx.x * blockDim.x + threadIdx.x) * 4;
  if (i + 3 < n) {
    out[i + 0] = (__bf16)in[i + 0];
    out[i + 1] = (__bf16)in[i + 1];
    out[i + 2] = (__bf16)in[i + 2];
    out[i + 3] = (__bf16)in[i + 3];
  }
}

// Wt[n][k] = (bf16) W[k][n]
__global__ void transpose_w_kernel(const float* __restrict__ W,
                                   __bf16* __restrict__ Wt) {
  int idx = blockIdx.x * blockDim.x + threadIdx.x;
  int k = idx >> 10;
  int n = idx & 1023;
  Wt[(size_t)n * DMODEL + k] = (__bf16)W[idx];
}

// ---- tiled WMMA GEMM, double-buffered async staging ------------------------
// C[M,1024] = A[M,1024] @ Bt^T + bias ; block tile 128x64, BK=32, 8 waves.
// mode 0: fp32 out[M,1024]; 1: bf16 [B,H,S,HD]; 2: bf16 [B,H,HD,S]
__global__ void gemm_wmma_kernel(const __bf16* __restrict__ A,
                                 const __bf16* __restrict__ Bt,
                                 const float* __restrict__ bias,
                                 float* __restrict__ outF,
                                 __bf16* __restrict__ outB, int mode) {
  __shared__ __bf16 As[2][128][40];  // 80B row stride (16B aligned)
  __shared__ __bf16 Bs[2][64][40];

  const int tid  = threadIdx.x;
  const int wid  = tid >> 5;
  const int lane = tid & 31;
  const int half = lane >> 4;
  const int lr   = lane & 15;
  const int wr   = wid >> 1;
  const int wc   = wid & 1;

  const int m0 = blockIdx.x * 128;
  const int n0 = blockIdx.y * 64;

  // per-thread staging coordinates
  const int arow0 = tid >> 2,        acol = (tid & 3) * 8;          // +  i*64 rows
  const int brow  = tid >> 2;        const int bcol = (tid & 3) * 8;

  v8f acc[2][2];
#pragma unroll
  for (int i = 0; i < 2; ++i)
#pragma unroll
    for (int j = 0; j < 2; ++j) acc[i][j] = (v8f)0.0f;

  // prologue: stage k-slab 0 into buffer 0
#pragma unroll
  for (int i = 0; i < 2; ++i)
    async_copy16(&As[0][arow0 + i * 64][acol],
                 &A[(size_t)(m0 + arow0 + i * 64) * DMODEL + acol]);
  async_copy16(&Bs[0][brow][bcol], &Bt[(size_t)(n0 + brow) * DMODEL + bcol]);

  for (int it = 0; it < DMODEL / 32; ++it) {
    const int cur = it & 1, nxt = cur ^ 1;
    const int knext = (it < DMODEL / 32 - 1) ? (it + 1) * 32 : 0;  // clamp last
    // prefetch next slab into other buffer (its readers finished last barrier)
#pragma unroll
    for (int i = 0; i < 2; ++i)
      async_copy16(&As[nxt][arow0 + i * 64][acol],
                   &A[(size_t)(m0 + arow0 + i * 64) * DMODEL + knext + acol]);
    async_copy16(&Bs[nxt][brow][bcol],
                 &Bt[(size_t)(n0 + brow) * DMODEL + knext + bcol]);
    // wait for current buffer's 3 copies (allow the 3 just issued to fly)
    WAIT_ASYNC(0x3);
    __syncthreads();

    v16bf a0 = load_frag(&As[cur][wr * 32 + lr][0], half);
    v16bf a1 = load_frag(&As[cur][wr * 32 + 16 + lr][0], half);
    v16bf b0 = load_frag(&Bs[cur][wc * 32 + lr][0], half);
    v16bf b1 = load_frag(&Bs[cur][wc * 32 + 16 + lr][0], half);
    acc[0][0] = wmma_bf16(a0, b0, acc[0][0]);
    acc[0][1] = wmma_bf16(a0, b1, acc[0][1]);
    acc[1][0] = wmma_bf16(a1, b0, acc[1][0]);
    acc[1][1] = wmma_bf16(a1, b1, acc[1][1]);
    __syncthreads();
  }

  // epilogue: C layout -> row M = r + 8*half, col N = lane&15
#pragma unroll
  for (int mt = 0; mt < 2; ++mt) {
#pragma unroll
    for (int nt = 0; nt < 2; ++nt) {
      int n = n0 + wc * 32 + nt * 16 + lr;
      float bn = bias[n];
#pragma unroll
      for (int r = 0; r < 8; ++r) {
        int m = m0 + wr * 32 + mt * 16 + r + 8 * half;
        float cv = acc[mt][nt][r] + bn;
        if (mode == 0) {
          outF[(size_t)m * DMODEL + n] = cv;
        } else {
          int b = m >> 11, s = m & 2047;
          int h = n >> 6, dd = n & 63;
          if (mode == 1)
            outB[((size_t)(b * NHEAD + h) * SEQ + s) * HDIM + dd] = (__bf16)cv;
          else
            outB[((size_t)(b * NHEAD + h) * HDIM + dd) * SEQ + s] = (__bf16)cv;
        }
      }
    }
  }
}

// ---- flash attention, double-buffered async K/V staging --------------------
__global__ void flash_attn_kernel(const __bf16* __restrict__ Q,
                                  const __bf16* __restrict__ K,
                                  const __bf16* __restrict__ Vt,
                                  __bf16* __restrict__ ctxb) {
  __shared__ __bf16 Ks[2][64][72];    // 64 keys x 64 d
  __shared__ __bf16 Vts[2][64][72];   // 64 d x 64 keys
  __shared__ __bf16 Ps[8][16][72];    // per-wave P scratch 16 x 64

  const int tid  = threadIdx.x;
  const int wid  = tid >> 5;
  const int lane = tid & 31;
  const int half = lane >> 4;
  const int lr   = lane & 15;

  const int bh    = blockIdx.y;
  const int qrow0 = blockIdx.x * 128 + wid * 16;

  const __bf16* Qg  = Q  + (size_t)bh * SEQ * HDIM;
  const __bf16* Kg  = K  + (size_t)bh * SEQ * HDIM;
  const __bf16* Vtg = Vt + (size_t)bh * HDIM * SEQ;

  const __bf16* qrp = Qg + (size_t)(qrow0 + lr) * HDIM;
  v16bf aq0 = load_frag(qrp, half);
  v16bf aq1 = load_frag(qrp + 32, half);

  v8f ctx[4];
#pragma unroll
  for (int i = 0; i < 4; ++i) ctx[i] = (v8f)0.0f;
  float mi[8], li[8];
#pragma unroll
  for (int r = 0; r < 8; ++r) { mi[r] = -3.0e38f; li[r] = 0.0f; }

  __bf16* Pw = &Ps[wid][0][0];

  // staging coords: 512 16B chunks per tile, 2 per thread
  const int srow0 = tid >> 3;               // + i*32 rows
  const int se0   = (tid & 7) * 8;

  // prologue: chunk 0 -> buffer 0 (4 async copies per thread)
#pragma unroll
  for (int i = 0; i < 2; ++i) {
    int row = srow0 + i * 32;
    async_copy16(&Ks[0][row][se0],  &Kg[(size_t)row * HDIM + se0]);
    async_copy16(&Vts[0][row][se0], &Vtg[(size_t)row * SEQ + se0]);
  }

  for (int kc = 0; kc < SEQ / 64; ++kc) {
    const int cur = kc & 1, nxt = cur ^ 1;
    const int kbn = (kc < SEQ / 64 - 1) ? (kc + 1) * 64 : 0;  // clamp last
#pragma unroll
    for (int i = 0; i < 2; ++i) {
      int row = srow0 + i * 32;
      async_copy16(&Ks[nxt][row][se0],  &Kg[(size_t)(kbn + row) * HDIM + se0]);
      async_copy16(&Vts[nxt][row][se0], &Vtg[(size_t)row * SEQ + kbn + se0]);
    }
    WAIT_ASYNC(0x4);   // current buffer's 4 copies done; 4 new in flight
    __syncthreads();

    // 4 key sub-tiles of 16
#pragma unroll
    for (int kt16 = 0; kt16 < 4; ++kt16) {
      v8f s = (v8f)0.0f;
      v16bf kb0 = load_frag(&Ks[cur][kt16 * 16 + lr][0], half);
      s = wmma_bf16(aq0, kb0, s);
      v16bf kb1 = load_frag(&Ks[cur][kt16 * 16 + lr][32], half);
      s = wmma_bf16(aq1, kb1, s);

#pragma unroll
      for (int r = 0; r < 8; ++r) {
        float sv = s[r] * 0.125f;  // 1/sqrt(64)
        float mx = sv;
        mx = fmaxf(mx, SWZ_XOR(mx, 1));
        mx = fmaxf(mx, SWZ_XOR(mx, 2));
        mx = fmaxf(mx, SWZ_XOR(mx, 4));
        mx = fmaxf(mx, SWZ_XOR(mx, 8));
        float mn = fmaxf(mi[r], mx);
        float al = __expf(mi[r] - mn);
        float p  = __expf(sv - mn);
        float rs = p;
        rs += SWZ_XOR(rs, 1);
        rs += SWZ_XOR(rs, 2);
        rs += SWZ_XOR(rs, 4);
        rs += SWZ_XOR(rs, 8);
        li[r] = li[r] * al + rs;
        mi[r] = mn;
        ctx[0][r] *= al; ctx[1][r] *= al; ctx[2][r] *= al; ctx[3][r] *= al;
        Pw[(r + 8 * half) * 72 + kt16 * 16 + lr] = (__bf16)p;
      }
    }
    WAIT_DS0();  // wave-private LDS: P stores feed fragment loads

    // ctx += P(16x64) @ V(64x64) with Bt = Vt[d][key]
#pragma unroll
    for (int kt = 0; kt < 2; ++kt) {
      v16bf pa = load_frag(&Ps[wid][lr][kt * 32], half);
#pragma unroll
      for (int nt = 0; nt < 4; ++nt) {
        v16bf bv = load_frag(&Vts[cur][nt * 16 + lr][kt * 32], half);
        ctx[nt] = wmma_bf16(pa, bv, ctx[nt]);
      }
    }
    __syncthreads();
  }

  const int b = bh >> 4, h = bh & 15;
  float inv[8];
#pragma unroll
  for (int r = 0; r < 8; ++r) inv[r] = 1.0f / li[r];
#pragma unroll
  for (int nt = 0; nt < 4; ++nt) {
    int dg = h * HDIM + nt * 16 + lr;
#pragma unroll
    for (int r = 0; r < 8; ++r) {
      int srow = qrow0 + r + 8 * half;
      ctxb[((size_t)(b * SEQ + srow)) * DMODEL + dg] = (__bf16)(ctx[nt][r] * inv[r]);
    }
  }
}

// ---- launcher --------------------------------------------------------------

extern "C" void kernel_launch(void* const* d_in, const int* in_sizes, int n_in,
                              void* d_out, int out_size, void* d_ws, size_t ws_size,
                              hipStream_t stream) {
  const float* x  = (const float*)d_in[0];
  const float* Wq = (const float*)d_in[1];
  const float* bq = (const float*)d_in[2];
  const float* Wk = (const float*)d_in[3];
  const float* bk = (const float*)d_in[4];
  const float* Wv = (const float*)d_in[5];
  const float* bv = (const float*)d_in[6];
  const float* Wo = (const float*)d_in[7];
  const float* bo = (const float*)d_in[8];

  char* ws = (char*)d_ws;
  const size_t MB = 1024 * 1024;
  __bf16* xb   = (__bf16*)(ws + 0);
  __bf16* Wqt  = (__bf16*)(ws + 8 * MB);
  __bf16* Wkt  = (__bf16*)(ws + 10 * MB);
  __bf16* Wvt  = (__bf16*)(ws + 12 * MB);
  __bf16* Wot  = (__bf16*)(ws + 14 * MB);
  __bf16* Qb   = (__bf16*)(ws + 16 * MB);
  __bf16* Kb   = (__bf16*)(ws + 24 * MB);
  __bf16* Vtb  = (__bf16*)(ws + 32 * MB);
  __bf16* ctxb = (__bf16*)(ws + 40 * MB);

  cvt_f32_bf16_kernel<<<NTOK * DMODEL / (256 * 4), 256, 0, stream>>>(
      x, xb, NTOK * DMODEL);
  transpose_w_kernel<<<DMODEL * DMODEL / 256, 256, 0, stream>>>(Wq, Wqt);
  transpose_w_kernel<<<DMODEL * DMODEL / 256, 256, 0, stream>>>(Wk, Wkt);
  transpose_w_kernel<<<DMODEL * DMODEL / 256, 256, 0, stream>>>(Wv, Wvt);
  transpose_w_kernel<<<DMODEL * DMODEL / 256, 256, 0, stream>>>(Wo, Wot);

  dim3 ggrid(NTOK / 128, DMODEL / 64);
  gemm_wmma_kernel<<<ggrid, 256, 0, stream>>>(xb, Wqt, bq, nullptr, Qb, 1);
  gemm_wmma_kernel<<<ggrid, 256, 0, stream>>>(xb, Wkt, bk, nullptr, Kb, 1);
  gemm_wmma_kernel<<<ggrid, 256, 0, stream>>>(xb, Wvt, bv, nullptr, Vtb, 2);

  dim3 agrid(SEQ / 128, 2 * NHEAD);
  flash_attn_kernel<<<agrid, 256, 0, stream>>>(Qb, Kb, Vtb, ctxb);

  gemm_wmma_kernel<<<ggrid, 256, 0, stream>>>(ctxb, Wot, bo, (float*)d_out,
                                              nullptr, 0);
}
